// MutualRec_67396626809064
// MI455X (gfx1250) — matrix operand order repaired and checked by reading, not compiled, not run on hardware
//
#include <hip/hip_runtime.h>

typedef __attribute__((ext_vector_type(16))) __bf16 v16bf;
typedef __attribute__((ext_vector_type(8)))  float  v8f;

#define DV 64

// ---------- helpers ----------
__device__ __forceinline__ unsigned short f2bf(float f) {
  unsigned u = __float_as_uint(f);
  u += 0x7FFFu + ((u >> 16) & 1u);      // round-to-nearest-even
  return (unsigned short)(u >> 16);
}
// order-preserving key for float atomicMax on unsigned
__device__ __forceinline__ unsigned okey(float f) {
  unsigned u = __float_as_uint(f);
  return (u & 0x80000000u) ? ~u : (u | 0x80000000u);
}
__device__ __forceinline__ float unkey(unsigned k) {
  unsigned u = (k & 0x80000000u) ? (k ^ 0x80000000u) : ~k;
  return __uint_as_float(u);
}

union FragAB {
  v16bf v;
  uint4 q[2];
  uint2 d[4];
  unsigned short us[16];
};

// ---------- generic fill ----------
__global__ void fill_u32(unsigned* __restrict__ p, unsigned v, int n) {
  int i = blockIdx.x * 256 + threadIdx.x;
  if (i < n) p[i] = v;
}

// ---------- fp32 -> bf16 convert ----------
__global__ void cvt_bf16(const float* __restrict__ x, unsigned short* __restrict__ y, int n) {
  int i = blockIdx.x * 256 + threadIdx.x;
  if (i < n) y[i] = f2bf(x[i]);
}

// ---------- big NT GEMM: C[M,N] = A[M,64] * B[N,64]^T (bf16 in, fp32 out) ----------
// block = 8 waves; each wave computes a 32x32 tile; block covers 32 rows x 256 cols.
// NSTAT: compile-time N (8192) so output stores fold to base + immediate IOFFSET;
// 0 selects the runtime-N fallback. Output stores are non-temporal (write-once,
// never re-read) to stream past L2 and keep it clean for the resident operands.
template <int NSTAT>
__global__ __launch_bounds__(256) void gemm_nt_bf16(
    const unsigned short* __restrict__ A, const unsigned short* __restrict__ B,
    float* __restrict__ C, int M, int N) {
  const int Nn = NSTAT ? NSTAT : N;
  const int lane = threadIdx.x & 31;
  const int wave = threadIdx.x >> 5;
  const int row_base = blockIdx.y * 32;
  const int col_base = blockIdx.x * 256 + wave * 32;
  const int lrow = lane & 15;
  const int hi = lane >> 4;

  v8f c00 = {}, c01 = {}, c10 = {}, c11 = {};
#pragma unroll
  for (int k0 = 0; k0 < DV; k0 += 32) {
    FragAB a0, a1, b0, b1;
    // A fragments: 16x32 bf16, lane group split K by 0/8, elements 8..15 at K+16
    {
      int grp = k0 + (hi ? 8 : 0);
      const unsigned short* p0 = A + (size_t)(row_base + lrow) * DV + grp;
      a0.q[0] = *(const uint4*)p0;
      a0.q[1] = *(const uint4*)(p0 + 16);
      const unsigned short* p1 = A + (size_t)(row_base + 16 + lrow) * DV + grp;
      a1.q[0] = *(const uint4*)p1;
      a1.q[1] = *(const uint4*)(p1 + 16);
    }
    // B fragments: 32x16 bf16, col = lane%16 (a row of B since NT), K split 0/8, elements 8..15 at K+16
    {
      int kb = k0 + (hi ? 8 : 0);
      const unsigned short* p0 = B + (size_t)(col_base + lrow) * DV + kb;
      b0.q[0] = *(const uint4*)p0;
      b0.q[1] = *(const uint4*)(p0 + 16);
      const unsigned short* p1 = B + (size_t)(col_base + 16 + lrow) * DV + kb;
      b1.q[0] = *(const uint4*)p1;
      b1.q[1] = *(const uint4*)(p1 + 16);
    }
    c00 = __builtin_amdgcn_wmma_f32_16x16x32_bf16(false, a0.v, false, b0.v, (short)0, c00, false, false);
    c01 = __builtin_amdgcn_wmma_f32_16x16x32_bf16(false, a0.v, false, b1.v, (short)0, c01, false, false);
    c10 = __builtin_amdgcn_wmma_f32_16x16x32_bf16(false, a1.v, false, b0.v, (short)0, c10, false, false);
    c11 = __builtin_amdgcn_wmma_f32_16x16x32_bf16(false, a1.v, false, b1.v, (short)0, c11, false, false);
  }
  const int srow = row_base + (hi ? 8 : 0);
  const int scol = col_base + lrow;
  float* cp = C + (size_t)srow * Nn + scol;
#pragma unroll
  for (int r = 0; r < 8; ++r) {
    __builtin_nontemporal_store(c00[r], cp + (size_t)r * Nn);
    __builtin_nontemporal_store(c01[r], cp + (size_t)r * Nn + 16);
    __builtin_nontemporal_store(c10[r], cp + (size_t)(r + 16) * Nn);
    __builtin_nontemporal_store(c11[r], cp + (size_t)(r + 16) * Nn + 16);
  }
}

// ---------- node linear: Y[N,64] = concat(Xa,Xb,Xc)[N,K] @ W[K,64] + bias ----------
// W staged transposed in LDS as bf16; each wave computes 16 rows x 64 cols via WMMA.
__global__ __launch_bounds__(256) void node_linear(
    const float* __restrict__ Xa, const float* __restrict__ Xb, const float* __restrict__ Xc,
    int nparts, const float* __restrict__ W, const float* __restrict__ bias,
    float* __restrict__ Y, int N) {
  __shared__ __align__(16) unsigned short Wt[64][196];   // [col][k], padded
  const int K = nparts * 64;
  for (int idx = threadIdx.x; idx < K * 64; idx += 256) {
    int k = idx >> 6, n = idx & 63;
    Wt[n][k] = f2bf(W[idx]);
  }
  __syncthreads();

  const int lane = threadIdx.x & 31;
  const int wave = threadIdx.x >> 5;
  const int row_base = (blockIdx.x * 8 + wave) * 16;
  if (row_base >= N) return;
  const int lrow = lane & 15;
  const int hi = lane >> 4;

  v8f vz = {};
  v8f c[4];
  c[0] = vz; c[1] = vz; c[2] = vz; c[3] = vz;

  for (int k0 = 0; k0 < K; k0 += 32) {
    const float* X = (k0 < 64) ? Xa : ((k0 < 128) ? Xb : Xc);
    int grp = (k0 & 63) + (hi ? 8 : 0);
    const float* xp = X + (size_t)(row_base + lrow) * DV + grp;
    float4 f0 = *(const float4*)xp;
    float4 f1 = *(const float4*)(xp + 4);
    float4 f2 = *(const float4*)(xp + 16);
    float4 f3 = *(const float4*)(xp + 20);
    FragAB a;
    a.us[0] = f2bf(f0.x);  a.us[1] = f2bf(f0.y);  a.us[2] = f2bf(f0.z);  a.us[3] = f2bf(f0.w);
    a.us[4] = f2bf(f1.x);  a.us[5] = f2bf(f1.y);  a.us[6] = f2bf(f1.z);  a.us[7] = f2bf(f1.w);
    a.us[8] = f2bf(f2.x);  a.us[9] = f2bf(f2.y);  a.us[10] = f2bf(f2.z); a.us[11] = f2bf(f2.w);
    a.us[12] = f2bf(f3.x); a.us[13] = f2bf(f3.y); a.us[14] = f2bf(f3.z); a.us[15] = f2bf(f3.w);

    int kb = k0 + (hi ? 8 : 0);
#pragma unroll
    for (int nt = 0; nt < 4; ++nt) {
      FragAB b;
      const unsigned short* wp = &Wt[nt * 16 + lrow][kb];
      b.d[0] = *(const uint2*)wp;
      b.d[1] = *(const uint2*)(wp + 4);
      b.d[2] = *(const uint2*)(wp + 16);
      b.d[3] = *(const uint2*)(wp + 20);
      c[nt] = __builtin_amdgcn_wmma_f32_16x16x32_bf16(false, a.v, false, b.v, (short)0, c[nt], false, false);
    }
  }

  const int srow = row_base + (hi ? 8 : 0);
#pragma unroll
  for (int nt = 0; nt < 4; ++nt) {
    int col = nt * 16 + lrow;
    float bv = bias[col];
#pragma unroll
    for (int r = 0; r < 8; ++r)
      Y[(size_t)(srow + r) * DV + col] = c[nt][r] + bv;
  }
}

// ---------- GATv2 edge kernels (wave per edge) ----------
__global__ void edge_score(const float* __restrict__ el, const float* __restrict__ er,
                           const float* __restrict__ av, const int* __restrict__ src,
                           const int* __restrict__ dst, int E,
                           float* __restrict__ s_out, unsigned* __restrict__ mkeys) {
  int e = blockIdx.x * 8 + (threadIdx.x >> 5);
  if (e >= E) return;
  int lane = threadIdx.x & 31;
  int uu = src[e], vv = dst[e];
  const float* pl = el + (size_t)uu * DV;
  const float* pr = er + (size_t)vv * DV;
  float t0 = pl[lane] + pr[lane];
  float t1 = pl[lane + 32] + pr[lane + 32];
  t0 = t0 > 0.f ? t0 : 0.2f * t0;
  t1 = t1 > 0.f ? t1 : 0.2f * t1;
  float acc = t0 * av[lane] + t1 * av[lane + 32];
  for (int o = 16; o; o >>= 1) acc += __shfl_xor(acc, o, 32);
  if (lane == 0) {
    s_out[e] = acc;
    atomicMax(mkeys + vv, okey(acc));
  }
}

__global__ void edge_exp(const float* __restrict__ s_in, const int* __restrict__ dst,
                         const unsigned* __restrict__ mkeys, int E,
                         float* __restrict__ w_out, float* __restrict__ z) {
  int e = blockIdx.x * 256 + threadIdx.x;
  if (e >= E) return;
  int v = dst[e];
  float w = __expf(s_in[e] - unkey(mkeys[v]));
  w_out[e] = w;
  atomicAdd(z + v, w);
}

__global__ void edge_scatter(const float* __restrict__ el, const int* __restrict__ src,
                             const int* __restrict__ dst, const float* __restrict__ w_e,
                             const float* __restrict__ z, int E, float* __restrict__ out) {
  int e = blockIdx.x * 8 + (threadIdx.x >> 5);
  if (e >= E) return;
  int lane = threadIdx.x & 31;
  int uu = src[e], vv = dst[e];
  float alpha = w_e[e] / z[vv];
  const float* pl = el + (size_t)uu * DV;
  float* po = out + (size_t)vv * DV;
  atomicAdd(po + lane,      alpha * pl[lane]);
  atomicAdd(po + lane + 32, alpha * pl[lane + 32]);
}

// ---------- Chebyshev propagation kernels ----------
__global__ void edge_copy_scatter(const float* __restrict__ xn, const int* __restrict__ src,
                                  const int* __restrict__ dst, int E, float* __restrict__ acc) {
  int e = blockIdx.x * 8 + (threadIdx.x >> 5);
  if (e >= E) return;
  int lane = threadIdx.x & 31;
  int uu = src[e], vv = dst[e];
  const float* pl = xn + (size_t)uu * DV;
  float* po = acc + (size_t)vv * DV;
  atomicAdd(po + lane,      pl[lane]);
  atomicAdd(po + lane + 32, pl[lane + 32]);
}

__global__ void deg_kernel(const int* __restrict__ dst, int E, float* __restrict__ deg) {
  int e = blockIdx.x * 256 + threadIdx.x;
  if (e < E) atomicAdd(deg + dst[e], 1.0f);
}

__global__ void norm_kernel(const float* __restrict__ deg, int N, float* __restrict__ nrm) {
  int i = blockIdx.x * 256 + threadIdx.x;
  if (i < N) nrm[i] = rsqrtf(fmaxf(deg[i], 1.0f));
}

__global__ void scale_rows(const float* __restrict__ x, const float* __restrict__ nrm,
                           float* __restrict__ y, int n) {
  int i = blockIdx.x * 256 + threadIdx.x;
  if (i < n) y[i] = x[i] * nrm[i >> 6];
}

__global__ void cheb_x1(const float* __restrict__ acc, const float* __restrict__ nrm,
                        const float* __restrict__ X0, const float* __restrict__ lam,
                        float* __restrict__ X1, int n) {
  int i = blockIdx.x * 256 + threadIdx.x;
  if (i >= n) return;
  float re = 2.0f / lam[0];
  X1[i] = -re * acc[i] * nrm[i >> 6] + (re - 1.0f) * X0[i];
}

__global__ void cheb_x2(const float* __restrict__ acc, const float* __restrict__ nrm,
                        const float* __restrict__ X1, const float* __restrict__ X0,
                        const float* __restrict__ lam, float* __restrict__ X2, int n) {
  int i = blockIdx.x * 256 + threadIdx.x;
  if (i >= n) return;
  float re = 2.0f / lam[0];
  X2[i] = -2.0f * re * acc[i] * nrm[i >> 6] + 2.0f * (re - 1.0f) * X1[i] - X0[i];
}

// ---------- mutualistic layer (wave per row; row softmax over 64 dims) ----------
__global__ void mutual_kernel(const float* __restrict__ hP, const float* __restrict__ hS,
                              float* __restrict__ hmP, float* __restrict__ hmS, int N) {
  int row = blockIdx.x * 8 + (threadIdx.x >> 5);
  if (row >= N) return;
  int lane = threadIdx.x & 31;
  const float* pp = hP + (size_t)row * DV;
  const float* ps = hS + (size_t)row * DV;
  float p0 = pp[lane], p1 = pp[lane + 32];
  float s0 = ps[lane], s1 = ps[lane + 32];
  float mp = fmaxf(p0, p1);
  for (int o = 16; o; o >>= 1) mp = fmaxf(mp, __shfl_xor(mp, o, 32));
  float ep0 = __expf(p0 - mp), ep1 = __expf(p1 - mp);
  float sp = ep0 + ep1;
  for (int o = 16; o; o >>= 1) sp += __shfl_xor(sp, o, 32);
  float msx = fmaxf(s0, s1);
  for (int o = 16; o; o >>= 1) msx = fmaxf(msx, __shfl_xor(msx, o, 32));
  float es0 = __expf(s0 - msx), es1 = __expf(s1 - msx);
  float ss = es0 + es1;
  for (int o = 16; o; o >>= 1) ss += __shfl_xor(ss, o, 32);
  float hm0 = p0 * s0, hm1 = p1 * s1;
  float* oP = hmP + (size_t)row * DV;
  float* oS = hmS + (size_t)row * DV;
  oP[lane]      = hm0 * (ep0 / sp);
  oP[lane + 32] = hm1 * (ep1 / sp);
  oS[lane]      = hm0 * (es0 / ss);
  oS[lane + 32] = hm1 * (es1 / ss);
}

// ---------- host orchestration ----------
extern "C" void kernel_launch(void* const* d_in, const int* in_sizes, int n_in,
                              void* d_out, int out_size, void* d_ws, size_t ws_size,
                              hipStream_t stream) {
  const float* u  = (const float*)d_in[0];
  const float* it = (const float*)d_in[1];
  const float *gWs[5], *gbs[5], *gWd[5], *gbd[5], *ga[5];
  for (int i = 0; i < 5; ++i) {
    int b = 2 + i * 5;
    gWs[i] = (const float*)d_in[b];     gbs[i] = (const float*)d_in[b + 1];
    gWd[i] = (const float*)d_in[b + 2]; gbd[i] = (const float*)d_in[b + 3];
    ga[i]  = (const float*)d_in[b + 4];
  }
  const float* cheb_W = (const float*)d_in[27]; const float* cheb_b = (const float*)d_in[28];
  const float* out_W  = (const float*)d_in[29]; const float* out_b  = (const float*)d_in[30];
  const float* mc_W   = (const float*)d_in[31]; const float* mc_b   = (const float*)d_in[32];
  const float* ms_W   = (const float*)d_in[33]; const float* ms_b   = (const float*)d_in[34];
  const float* pp_W   = (const float*)d_in[35]; const float* pp_b   = (const float*)d_in[36];
  const float* ps_W   = (const float*)d_in[37]; const float* ps_b   = (const float*)d_in[38];
  const float* lam    = (const float*)d_in[39];
  const int* rate_src   = (const int*)d_in[40];
  const int* rate_dst   = (const int*)d_in[41];
  const int* friend_src = (const int*)d_in[42];
  const int* friend_dst = (const int*)d_in[43];

  const int NU = in_sizes[0] / DV;
  const int NI = in_sizes[1] / DV;
  const int ER = in_sizes[40];
  const int EF = in_sizes[42];
  const int NMAX = (NU > NI) ? NU : NI;
  const int NF = NU * DV;  // per-node-matrix element count (NU==NI here)

  // workspace carve
  char* wsp = (char*)d_ws;
  size_t off = 0;
  auto carve = [&](size_t bytes) {
    void* p = wsp + off;
    off += (bytes + 255) & ~(size_t)255;
    return p;
  };
  const size_t NB = (size_t)NMAX * DV * sizeof(float);
  float* el   = (float*)carve(NB);
  float* er   = (float*)carve(NB);
  float* b2   = (float*)carve(NB);  // h1_item -> user_pref -> h_new_S
  float* b3   = (float*)carve(NB);  // h2_user -> ch
  float* b4   = (float*)carve(NB);  // item_inf -> user_social
  float* b5   = (float*)carve(NB);  // social_it -> h_uP
  float* b6   = (float*)carve(NB);  // xn -> h_new_P
  float* b7   = (float*)carve(NB);  // X1 -> h_uS
  float* b8   = (float*)carve(NB);  // X2 -> h_mP
  float* b9   = (float*)carve(NB);  // acc -> h_mS
  float* s_e  = (float*)carve((size_t)ER * sizeof(float));
  float* w_e  = (float*)carve((size_t)ER * sizeof(float));
  unsigned* mkeys = (unsigned*)carve((size_t)NMAX * sizeof(unsigned));
  float* zb    = (float*)carve((size_t)NMAX * sizeof(float));
  float* degv  = (float*)carve((size_t)NU * sizeof(float));
  float* nrmv  = (float*)carve((size_t)NU * sizeof(float));
  unsigned short* Pb = (unsigned short*)carve((size_t)NU * DV * sizeof(unsigned short));
  unsigned short* Sb = (unsigned short*)carve((size_t)NU * DV * sizeof(unsigned short));
  unsigned short* Ib = (unsigned short*)carve((size_t)NI * DV * sizeof(unsigned short));
  unsigned short* Ub = (unsigned short*)carve((size_t)NU * DV * sizeof(unsigned short));
  (void)ws_size; (void)n_in; (void)out_size;

  auto fill = [&](void* p, unsigned v, int n) {
    fill_u32<<<(n + 255) / 256, 256, 0, stream>>>((unsigned*)p, v, n);
  };

  auto launch_gemm = [&](const unsigned short* A, const unsigned short* B, float* C,
                         int M, int N) {
    dim3 g(N / 256, M / 32);
    if (N == 8192)
      gemm_nt_bf16<8192><<<g, 256, 0, stream>>>(A, B, C, M, N);
    else
      gemm_nt_bf16<0><<<g, 256, 0, stream>>>(A, B, C, M, N);
  };

  auto run_gat = [&](const float* xs, const float* xd, const int* src, const int* dst,
                     int Ns, int Nd, int E,
                     const float* Ws, const float* bs, const float* Wd, const float* bd,
                     const float* av, float* outp) {
    node_linear<<<Ns / 128, 256, 0, stream>>>(xs, xs, xs, 1, Ws, bs, el, Ns);
    node_linear<<<Nd / 128, 256, 0, stream>>>(xd, xd, xd, 1, Wd, bd, er, Nd);
    fill(mkeys, 0u, Nd);
    fill(zb, 0u, Nd);
    fill(outp, 0u, Nd * DV);
    edge_score<<<(E + 7) / 8, 256, 0, stream>>>(el, er, av, src, dst, E, s_e, mkeys);
    edge_exp<<<(E + 255) / 256, 256, 0, stream>>>(s_e, dst, mkeys, E, w_e, zb);
    edge_scatter<<<(E + 7) / 8, 256, 0, stream>>>(el, src, dst, w_e, zb, E, outp);
  };

  // ---- spatial attention pipeline ----
  run_gat(u,  it, rate_src, rate_dst, NU, NI, ER, gWs[0], gbs[0], gWd[0], gbd[0], ga[0], b2); // h1_item
  run_gat(it, u,  rate_dst, rate_src, NI, NU, ER, gWs[1], gbs[1], gWd[1], gbd[1], ga[1], b3); // h2_user
  float* h1_item = b2;
  float* h2_user = b3;
  run_gat(h1_item, u, rate_dst, rate_src, NI, NU, ER, gWs[2], gbs[2], gWd[2], gbd[2], ga[2], b4); // item_inf
  run_gat(h2_user, u, friend_src, friend_dst, NU, NU, EF, gWs[3], gbs[3], gWd[3], gbd[3], ga[3], b5); // social_it
  float* user_pref = b2;  // h1_item dead
  node_linear<<<NU / 128, 256, 0, stream>>>(b4, b5, b5, 2, out_W, out_b, user_pref, NU);

  // ---- spectral branch: ChebConv(k=3) on social graph ----
  fill(degv, 0u, NU);
  deg_kernel<<<(EF + 255) / 256, 256, 0, stream>>>(friend_dst, EF, degv);
  norm_kernel<<<(NU + 255) / 256, 256, 0, stream>>>(degv, NU, nrmv);
  float* xn = b6; float* acc = b9; float* X1 = b7; float* X2 = b8;
  scale_rows<<<(NF + 255) / 256, 256, 0, stream>>>(u, nrmv, xn, NF);
  fill(acc, 0u, NF);
  edge_copy_scatter<<<(EF + 7) / 8, 256, 0, stream>>>(xn, friend_src, friend_dst, EF, acc);
  cheb_x1<<<(NF + 255) / 256, 256, 0, stream>>>(acc, nrmv, u, lam, X1, NF);
  scale_rows<<<(NF + 255) / 256, 256, 0, stream>>>(X1, nrmv, xn, NF);
  fill(acc, 0u, NF);
  edge_copy_scatter<<<(EF + 7) / 8, 256, 0, stream>>>(xn, friend_src, friend_dst, EF, acc);
  cheb_x2<<<(NF + 255) / 256, 256, 0, stream>>>(acc, nrmv, X1, u, lam, X2, NF);
  float* ch = b3;  // h2_user dead
  node_linear<<<NU / 128, 256, 0, stream>>>(u, X1, X2, 3, cheb_W, cheb_b, ch, NU);
  float* user_social = b4;  // item_inf dead
  run_gat(ch, ch, friend_src, friend_dst, NU, NU, EF, gWs[4], gbs[4], gWd[4], gbd[4], ga[4], user_social);

  // ---- mutualistic layer ----
  float* h_uP = b5;  // social_it dead
  float* h_uS = b7;  // X1 dead
  node_linear<<<NU / 128, 256, 0, stream>>>(user_pref, u, u, 2, mc_W, mc_b, h_uP, NU);
  node_linear<<<NU / 128, 256, 0, stream>>>(user_social, u, u, 2, ms_W, ms_b, h_uS, NU);
  float* h_mP = b8;  // X2 dead
  float* h_mS = b9;  // acc dead
  mutual_kernel<<<NU / 8, 256, 0, stream>>>(h_uP, h_uS, h_mP, h_mS, NU);

  // ---- prediction layer ----
  float* h_new_P = b6;  // xn dead
  float* h_new_S = b2;  // user_pref dead after mc
  node_linear<<<NU / 128, 256, 0, stream>>>(h_mP, h_uP, h_uP, 2, pp_W, pp_b, h_new_P, NU);
  node_linear<<<NU / 128, 256, 0, stream>>>(h_mS, h_uS, h_uS, 2, ps_W, ps_b, h_new_S, NU);

  // ---- score matrices via bf16 WMMA (L2-resident operands, store-bound) ----
  cvt_bf16<<<(NF + 255) / 256, 256, 0, stream>>>(h_new_P, Pb, NF);
  cvt_bf16<<<(NF + 255) / 256, 256, 0, stream>>>(h_new_S, Sb, NF);
  cvt_bf16<<<(NI * DV + 255) / 256, 256, 0, stream>>>(it, Ib, NI * DV);
  cvt_bf16<<<(NF + 255) / 256, 256, 0, stream>>>(u, Ub, NF);

  float* r_hat = (float*)d_out;
  float* s_hat = r_hat + (size_t)NU * NI;
  launch_gemm(Pb, Ib, r_hat, NU, NI);
  launch_gemm(Sb, Ub, s_hat, NU, NU);
}
